// DeepConvWeigthNet_32856499814953
// MI455X (gfx1250) — compile-verified, matrix operand order checked by reading, not run on hardware
//
#include <hip/hip_runtime.h>
#include <hip/hip_bf16.h>
#include <stdint.h>

// ---------------------------------------------------------------------------
// MI455X (gfx1250) implementation.
//  - conv2/conv3/head convs: implicit-GEMM via v_wmma_f32_16x16x32_f16,
//    9 tap-GEMMs over NHWC f16 activations with a zero halo (SAME padding).
//  - weights staged into LDS with the Tensor Data Mover (tensor_load_to_lds
//    + s_wait_tensorcnt) -- CDNA5 async-tensor path (6-arg clang-23 builtin).
//  - box blurs: separable sliding-window (O(1)/pixel), softmax fused into the
//    weighted-combine kernel.
// ---------------------------------------------------------------------------

typedef __attribute__((ext_vector_type(16))) _Float16 v16h;
typedef __attribute__((ext_vector_type(8)))  float    v8f;
typedef unsigned int u32x4 __attribute__((ext_vector_type(4)));
typedef int          i32x8 __attribute__((ext_vector_type(8)));
typedef int          i32x4 __attribute__((ext_vector_type(4)));

#define BB 8
#define HH 512
#define WW 512
#define HP 514
#define WP 514

// ---------------------------------------------------------------------------
// Weight re-layout: OIHW f32 -> [tap][cout][cin] f16 (B-fragment friendly)
// ---------------------------------------------------------------------------
__global__ void prep_weights(const float* __restrict__ w2, const float* __restrict__ w3,
                             const float* __restrict__ hw1, const float* __restrict__ hw2,
                             const float* __restrict__ hw3,
                             _Float16* __restrict__ wt2h, _Float16* __restrict__ wt3h,
                             _Float16* __restrict__ h1h, _Float16* __restrict__ h2h,
                             _Float16* __restrict__ h3h) {
    int id = blockIdx.x * blockDim.x + threadIdx.x;
    if (id < 9 * 64 * 32) {                       // conv2: [9][64][32]
        int i = id % 32, o = (id / 32) % 64, t = id / (32 * 64);
        wt2h[id] = (_Float16)w2[(o * 32 + i) * 9 + t];
    }
    int id2 = id - 9 * 64 * 32;
    if (id2 >= 0 && id2 < 9 * 32 * 64) {          // conv3: [9][32][64]
        int i = id2 % 64, o = (id2 / 64) % 32, t = id2 / (64 * 32);
        wt3h[id2] = (_Float16)w3[(o * 64 + i) * 9 + t];
    }
    int id3 = id - 2 * 9 * 64 * 32;
    if (id3 >= 0 && id3 < 3 * 9 * 16 * 32) {      // heads: [9][16][32], cout padded 4->16
        int h = id3 / (9 * 16 * 32); int r = id3 % (9 * 16 * 32);
        int i = r % 32, o = (r / 32) % 16, t = r / (32 * 16);
        const float* hw = (h == 0) ? hw1 : (h == 1) ? hw2 : hw3;
        _Float16* dst = (h == 0) ? h1h : (h == 1) ? h2h : h3h;
        dst[r] = (_Float16)((o < 4) ? hw[(o * 32 + i) * 9 + t] : 0.f);
    }
}

// ---------------------------------------------------------------------------
// conv1: 1->32, fp32 in -> f16 NHWC padded out, fused bias + PReLU (tiny K)
// ---------------------------------------------------------------------------
__global__ __launch_bounds__(256) void conv1_prelu(const float* __restrict__ x,
                                                   const float* __restrict__ w1,
                                                   const float* __restrict__ b1,
                                                   const float* __restrict__ a1,
                                                   _Float16* __restrict__ out) {
    __shared__ float sw[288 + 32];
    for (int i = threadIdx.x; i < 320; i += blockDim.x) sw[i] = (i < 288) ? w1[i] : b1[i - 288];
    __syncthreads();
    long gid = (long)blockIdx.x * blockDim.x + threadIdx.x;
    if (gid >= (long)BB * HH * WW) return;
    int xx = (int)(gid % WW); int yy = (int)((gid / WW) % HH); int b = (int)(gid / ((long)HH * WW));
    float alpha = a1[0];
    float taps[9];
#pragma unroll
    for (int ky = 0; ky < 3; ky++)
#pragma unroll
        for (int kx = 0; kx < 3; kx++) {
            int iy = yy + ky - 1, ix = xx + kx - 1;
            taps[ky * 3 + kx] = (iy >= 0 && iy < HH && ix >= 0 && ix < WW)
                                    ? x[((long)b * HH + iy) * WW + ix] : 0.f;
        }
    _Float16* op = out + (((long)b * HP + yy + 1) * WP + xx + 1) * 32;
    for (int c = 0; c < 32; c++) {
        float s = sw[288 + c];
#pragma unroll
        for (int t = 0; t < 9; t++) s += sw[c * 9 + t] * taps[t];
        s = (s >= 0.f) ? s : alpha * s;
        op[c] = (_Float16)s;
    }
}

// ---------------------------------------------------------------------------
// 3x3 conv as 9 tap-GEMMs: v_wmma_f32_16x16x32_f16. Wave = 16 px x COUT.
// Weights DMA'd to LDS via Tensor Data Mover.
// ---------------------------------------------------------------------------
template <int CIN, int COUT, bool HEAD>
__global__ __launch_bounds__(256) void conv3x3_wmma(const _Float16* __restrict__ in,
                                                    const _Float16* __restrict__ wt,
                                                    const float* __restrict__ bias,
                                                    const float* __restrict__ alphap,
                                                    _Float16* __restrict__ out,
                                                    float* __restrict__ yout) {
    constexpr int NT = COUT / 16;
    constexpr int WELEMS = 9 * COUT * CIN;
    __shared__ _Float16 smem[WELEMS];

    const int lane = threadIdx.x & 31;
    const int wave = threadIdx.x >> 5;
    const int b = blockIdx.z, y = blockIdx.y;
    const int x0 = blockIdx.x * 128 + wave * 16;

#if defined(__gfx1250__)
    if (wave == 0) {
        // 1-D TDM descriptor: whole weight block -> LDS, data_size = 8B units.
        const uint32_t n8 = (WELEMS * 2) / 8;
        uint32_t lds = (uint32_t)(uintptr_t)(void*)&smem[0];
        uint64_t ga = (uint64_t)(uintptr_t)wt;
        u32x4 g0;
        g0.x = 1u;                                       // count=1, user desc
        g0.y = lds;                                      // lds_addr
        g0.z = (uint32_t)ga;                             // global_addr[31:0]
        g0.w = ((uint32_t)(ga >> 32) & 0x01FFFFFFu) | (2u << 30);  // addr hi | type=2
        i32x8 g1;
        g1[0] = (3 << 16);                               // data_size=8B, no mask
        g1[1] = (int)((n8 & 0xFFFFu) << 16);             // tensor_dim0 lo
        g1[2] = (int)((n8 >> 16) | (1u << 16));          // tensor_dim0 hi | tensor_dim1=1
        g1[3] = (int)(n8 << 16);                         // tile_dim0 = n8
        g1[4] = 0;                                       // tile_dim1/2 unused
        g1[5] = (int)n8;                                 // tensor_dim0_stride
        g1[6] = 0; g1[7] = 0;
        i32x4 z4 = {0, 0, 0, 0};
        i32x8 z8 = {0, 0, 0, 0, 0, 0, 0, 0};
        __builtin_amdgcn_tensor_load_to_lds(g0, g1, z4, z4, z8, 0);
        __builtin_amdgcn_s_wait_tensorcnt(0);
    }
#else
    for (int i = threadIdx.x; i < WELEMS; i += blockDim.x) smem[i] = wt[i];
#endif
    __syncthreads();

    const int m = lane & 15;       // GEMM row (pixel) for A, column (cout) for B
    const int kh = lane >> 4;      // which K-half of 32 this lane carries
    const float alpha = alphap[0];

    v8f acc[NT];
#pragma unroll
    for (int n = 0; n < NT; n++) {
        float bv = HEAD ? ((m < 4) ? bias[m] : 0.f) : bias[n * 16 + m];
        v8f t;
#pragma unroll
        for (int r = 0; r < 8; r++) t[r] = bv;
        acc[n] = t;
    }

    const _Float16* ibase = in + (((long)b * HP + y) * WP + x0) * CIN;
    __builtin_prefetch(ibase + (long)3 * WP * CIN, 0, 1);  // next-row hint

#pragma unroll
    for (int ky = 0; ky < 3; ky++) {
#pragma unroll
        for (int kx = 0; kx < 3; kx++) {
            const int t = ky * 3 + kx;
#pragma unroll
            for (int ks = 0; ks < CIN / 32; ks++) {
                const _Float16* ap = ibase + (long)(ky * WP + kx + m) * CIN + ks * 32 + kh * 16;
                v16h a = *(const v16h*)ap;
#pragma unroll
                for (int n = 0; n < NT; n++) {
                    const _Float16* bp = smem + (size_t)(t * COUT + n * 16 + m) * CIN
                                              + ks * 32 + kh * 16;
                    v16h bf = *(const v16h*)bp;
#if defined(__gfx1250__)
                    acc[n] = __builtin_amdgcn_wmma_f32_16x16x32_f16(
                        false, a, false, bf, (short)0, acc[n], false, false);
#endif
                }
            }
        }
    }

    // D layout: lane L, VGPR r -> M = (L>>4)*8 + r, N = L&15
#pragma unroll
    for (int n = 0; n < NT; n++) {
#pragma unroll
        for (int r = 0; r < 8; r++) {
            float v = acc[n][r];
            const int mp = (lane >> 4) * 8 + r;
            const int c = n * 16 + (lane & 15);
            const int xo = x0 + mp;
            if (HEAD) {
                if (c < 4) yout[(((long)b * HH + y) * WW + xo) * 4 + c] = v;
            } else {
                v = (v >= 0.f) ? v : alpha * v;
                out[(((long)b * HP + (y + 1)) * WP + (xo + 1)) * COUT + c] = (_Float16)v;
            }
        }
    }
}

// ---------------------------------------------------------------------------
// Channel-attention: per-(head,b) channel means, then 4x4 MLP gates
// ---------------------------------------------------------------------------
__global__ __launch_bounds__(256) void head_mean(const float* __restrict__ yh1,
                                                 const float* __restrict__ yh2,
                                                 const float* __restrict__ yh3,
                                                 float* __restrict__ means) {
    int b = blockIdx.x, head = blockIdx.y;
    const float* yh = (head == 0) ? yh1 : (head == 1) ? yh2 : yh3;
    const float* p = yh + (long)b * HH * WW * 4;
    float s[4] = {0, 0, 0, 0};
    for (long i = threadIdx.x; i < (long)HH * WW; i += 256) {
#pragma unroll
        for (int c = 0; c < 4; c++) s[c] += p[i * 4 + c];
    }
    __shared__ float red[256][4];
#pragma unroll
    for (int c = 0; c < 4; c++) red[threadIdx.x][c] = s[c];
    __syncthreads();
    for (int off = 128; off > 0; off >>= 1) {
        if (threadIdx.x < (unsigned)off)
#pragma unroll
            for (int c = 0; c < 4; c++) red[threadIdx.x][c] += red[threadIdx.x + off][c];
        __syncthreads();
    }
    if (threadIdx.x == 0)
#pragma unroll
        for (int c = 0; c < 4; c++)
            means[(head * 8 + b) * 4 + c] = red[0][c] * (1.f / ((float)HH * (float)WW));
}

__global__ void ca_gates(const float* __restrict__ means,
                         const float* ca1a, const float* ca1b,
                         const float* ca2a, const float* ca2b,
                         const float* ca3a, const float* ca3b,
                         float* __restrict__ gates) {
    int id = threadIdx.x;
    if (id >= 24) return;
    int head = id >> 3, b = id & 7;
    const float* A = (head == 0) ? ca1a : (head == 1) ? ca2a : ca3a;
    const float* Bm = (head == 0) ? ca1b : (head == 1) ? ca2b : ca3b;
    const float* m = means + (head * 8 + b) * 4;
    float t[4];
#pragma unroll
    for (int c = 0; c < 4; c++) {
        float s = 0;
#pragma unroll
        for (int j = 0; j < 4; j++) s += A[c * 4 + j] * m[j];
        t[c] = fmaxf(s, 0.f);
    }
#pragma unroll
    for (int c = 0; c < 4; c++) {
        float s = 0;
#pragma unroll
        for (int j = 0; j < 4; j++) s += Bm[c * 4 + j] * t[j];
        gates[(head * 8 + b) * 4 + c] = 1.f / (1.f + __expf(-s));
    }
}

// ---------------------------------------------------------------------------
// Separable zero-padded box filters: O(1)/pixel sliding-window sums
// ---------------------------------------------------------------------------
__global__ void box_h(const float* __restrict__ src, float* __restrict__ dst,
                      int r, float invk) {
    int row = blockIdx.x * blockDim.x + threadIdx.x;  // B*H rows
    if (row >= BB * HH) return;
    const float* s = src + (long)row * WW;
    float* d = dst + (long)row * WW;
    float acc = 0.f;
    for (int j = 0; j <= r; j++) acc += s[j];
    for (int x = 0; x < WW; x++) {
        d[x] = acc * invk;
        int xb = x + r + 1, xa = x - r;
        acc += ((xb < WW) ? s[xb] : 0.f) - ((xa >= 0) ? s[xa] : 0.f);
    }
}

__global__ void box_v(const float* __restrict__ src, float* __restrict__ dst,
                      int r, float invk) {
    int id = blockIdx.x * blockDim.x + threadIdx.x;  // B*W columns (coalesced)
    if (id >= BB * WW) return;
    int b = id / WW, col = id % WW;
    const float* s = src + (long)b * HH * WW + col;
    float* d = dst + (long)b * HH * WW + col;
    float acc = 0.f;
    for (int j = 0; j <= r; j++) acc += s[(long)j * WW];
    for (int y = 0; y < HH; y++) {
        d[(long)y * WW] = acc * invk;
        int yb = y + r + 1, ya = y - r;
        acc += ((yb < HH) ? s[(long)yb * WW] : 0.f) - ((ya >= 0) ? s[(long)ya * WW] : 0.f);
    }
}

// softmax over 4 gated head channels fused with the weighted blur combine
__global__ __launch_bounds__(256) void combine(const float* __restrict__ yh,
                                               const float* __restrict__ gates,
                                               const float* __restrict__ s0,
                                               const float* __restrict__ b5,
                                               const float* __restrict__ b15,
                                               const float* __restrict__ b25,
                                               float* __restrict__ out) {
    long p = (long)blockIdx.x * blockDim.x + threadIdx.x;
    if (p >= (long)BB * HH * WW) return;
    int b = (int)(p / ((long)HH * WW));
    const float* g = gates + b * 4;
    float z[4];
#pragma unroll
    for (int c = 0; c < 4; c++) z[c] = yh[p * 4 + c] * g[c];
    float mx = fmaxf(fmaxf(z[0], z[1]), fmaxf(z[2], z[3]));
    float e0 = __expf(z[0] - mx), e1 = __expf(z[1] - mx);
    float e2 = __expf(z[2] - mx), e3 = __expf(z[3] - mx);
    float inv = 1.f / (e0 + e1 + e2 + e3);
    out[p] = (e0 * s0[p] + e1 * b5[p] + e2 * b15[p] + e3 * b25[p]) * inv;
}

// ---------------------------------------------------------------------------
// Host orchestration
// ---------------------------------------------------------------------------
extern "C" void kernel_launch(void* const* d_in, const int* in_sizes, int n_in,
                              void* d_out, int out_size, void* d_ws, size_t ws_size,
                              hipStream_t stream) {
    (void)in_sizes; (void)n_in; (void)out_size; (void)ws_size;
    const float* x  = (const float*)d_in[0];
    const float* w1 = (const float*)d_in[1];
    const float* b1 = (const float*)d_in[2];
    const float* a1 = (const float*)d_in[3];
    const float* w2 = (const float*)d_in[4];
    const float* b2 = (const float*)d_in[5];
    const float* a2 = (const float*)d_in[6];
    const float* w3 = (const float*)d_in[7];
    const float* b3 = (const float*)d_in[8];
    const float* a3 = (const float*)d_in[9];
    const float* hw1 = (const float*)d_in[10]; const float* hb1 = (const float*)d_in[11];
    const float* ca1a = (const float*)d_in[12]; const float* ca1b = (const float*)d_in[13];
    const float* hw2 = (const float*)d_in[14]; const float* hb2 = (const float*)d_in[15];
    const float* ca2a = (const float*)d_in[16]; const float* ca2b = (const float*)d_in[17];
    const float* hw3 = (const float*)d_in[18]; const float* hb3 = (const float*)d_in[19];
    const float* ca3a = (const float*)d_in[20]; const float* ca3b = (const float*)d_in[21];

    char* ws = (char*)d_ws;
    // Workspace layout (bytes). act3 reuses act1; head/blur buffers reuse act2.
    const size_t ACT1_SZ = (size_t)BB * HP * WP * 32 * 2;   // 135,268,352
    const size_t ACT2_SZ = (size_t)BB * HP * WP * 64 * 2;   // 270,536,704
    const size_t ACT1 = 0, ACT2 = ACT1_SZ, ACT3 = 0;
    const size_t YH = 4uLL * BB * HH * WW * 4;              // 33,554,432
    const size_t IMG = 4uLL * BB * HH * WW;                 // 8,388,608
    const size_t YH1 = ACT2, YH2 = YH1 + YH, YH3 = YH2 + YH;
    const size_t TMPB = YH3 + YH;
    const size_t B5 = TMPB + IMG, B15 = B5 + IMG, B25 = B15 + IMG;
    const size_t OUT1 = B25 + IMG, OUT2 = OUT1 + IMG;
    const size_t SMALL = ACT1_SZ + ACT2_SZ;
    const size_t WT2H = SMALL, WT3H = SMALL + 40960;
    const size_t H1H = SMALL + 81920, H2H = SMALL + 94208, H3H = SMALL + 106496;
    const size_t MEANS = SMALL + 118784, GATES = SMALL + 119296;

    _Float16* act1 = (_Float16*)(ws + ACT1);
    _Float16* act2 = (_Float16*)(ws + ACT2);
    _Float16* act3 = (_Float16*)(ws + ACT3);

    // Zero act halos (conv kernels only write interiors -> implicit SAME pad)
    (void)hipMemsetAsync(ws, 0, ACT1_SZ + ACT2_SZ, stream);

    prep_weights<<<dim3((2 * 9 * 64 * 32 + 3 * 9 * 16 * 32 + 255) / 256), 256, 0, stream>>>(
        w2, w3, hw1, hw2, hw3,
        (_Float16*)(ws + WT2H), (_Float16*)(ws + WT3H),
        (_Float16*)(ws + H1H), (_Float16*)(ws + H2H), (_Float16*)(ws + H3H));

    conv1_prelu<<<dim3((BB * HH * WW + 255) / 256), 256, 0, stream>>>(x, w1, b1, a1, act1);

    dim3 cgrid(WW / 128, HH, BB);
    conv3x3_wmma<32, 64, false><<<cgrid, 256, 0, stream>>>(
        act1, (_Float16*)(ws + WT2H), b2, a2, act2, nullptr);
    conv3x3_wmma<64, 32, false><<<cgrid, 256, 0, stream>>>(
        act2, (_Float16*)(ws + WT3H), b3, a3, act3, nullptr);

    conv3x3_wmma<32, 16, true><<<cgrid, 256, 0, stream>>>(
        act3, (_Float16*)(ws + H1H), hb1, a1, nullptr, (float*)(ws + YH1));
    conv3x3_wmma<32, 16, true><<<cgrid, 256, 0, stream>>>(
        act3, (_Float16*)(ws + H2H), hb2, a1, nullptr, (float*)(ws + YH2));
    conv3x3_wmma<32, 16, true><<<cgrid, 256, 0, stream>>>(
        act3, (_Float16*)(ws + H3H), hb3, a1, nullptr, (float*)(ws + YH3));

    head_mean<<<dim3(BB, 3), 256, 0, stream>>>(
        (const float*)(ws + YH1), (const float*)(ws + YH2), (const float*)(ws + YH3),
        (float*)(ws + MEANS));
    ca_gates<<<1, 32, 0, stream>>>((const float*)(ws + MEANS),
                                   ca1a, ca1b, ca2a, ca2b, ca3a, ca3b,
                                   (float*)(ws + GATES));

    const float* srcs[3] = {x, (const float*)(ws + OUT1), (const float*)(ws + OUT2)};
    float* dsts[3] = {(float*)(ws + OUT1), (float*)(ws + OUT2), (float*)d_out};
    const float* yhs[3] = {(const float*)(ws + YH1), (const float*)(ws + YH2),
                           (const float*)(ws + YH3)};
    float* tmp = (float*)(ws + TMPB);
    float* bb5 = (float*)(ws + B5); float* bb15 = (float*)(ws + B15);
    float* bb25 = (float*)(ws + B25);

    for (int i = 0; i < 3; i++) {
        box_h<<<16, 256, 0, stream>>>(srcs[i], tmp, 2, 1.f / 5.f);
        box_v<<<16, 256, 0, stream>>>(tmp, bb5, 2, 1.f / 5.f);
        box_h<<<16, 256, 0, stream>>>(srcs[i], tmp, 7, 1.f / 15.f);
        box_v<<<16, 256, 0, stream>>>(tmp, bb15, 7, 1.f / 15.f);
        box_h<<<16, 256, 0, stream>>>(srcs[i], tmp, 12, 1.f / 25.f);
        box_v<<<16, 256, 0, stream>>>(tmp, bb25, 12, 1.f / 25.f);
        combine<<<dim3((BB * HH * WW + 255) / 256), 256, 0, stream>>>(
            yhs[i], (const float*)(ws + GATES) + i * 32, srcs[i], bb5, bb15, bb25, dsts[i]);
    }
}
